// Gate_33930241638461
// MI455X (gfx1250) — matrix operand order repaired and checked by reading, not compiled, not run on hardware
//
#include <hip/hip_runtime.h>
#include <hip/hip_bf16.h>
#include <cstdint>

// ---- problem constants (match reference) ----
#define D_DIM   1024
#define E_DIM   64
#define TOK_BLK 128          // tokens per workgroup (8 waves x 16 tokens)
#define KPANEL  128          // K-panel staged in LDS as bf16
#define EPS_F   1e-9f

typedef __attribute__((ext_vector_type(16))) __bf16 v16bf;
typedef __attribute__((ext_vector_type(4)))  __bf16 v4bf;
typedef __attribute__((ext_vector_type(8)))  float  v8f;
typedef __attribute__((ext_vector_type(4)))  float  v4f;   // native vector for nontemporal builtins
typedef __attribute__((ext_vector_type(4)))  unsigned int v4u;

struct alignas(16) Bits256 { v4u lo; v4u hi; };   // 32 bytes -> v16bf

// LDS layout:
//   sW:      64 rows x (KPANEL + 8) bf16  (pad keeps ds_load_b128 bank-spread)
//   sScore:  TOK_BLK rows x 68 floats     (stride 68 -> half-wave groups hit
//                                          disjoint bank ranges on store)
#define WROW   (KPANEL + 8)      // 136 bf16 = 272B row stride
#define SROW   68

__global__ void __launch_bounds__(256)
router_gate_kernel(const float* __restrict__ x,
                   const float* __restrict__ W,
                   const float* __restrict__ b,
                   const float* __restrict__ u,
                   int*   __restrict__ idxOut,
                   float* __restrict__ logitsOut,
                   float* __restrict__ wOut,
                   int T)
{
    __shared__ __bf16 sW[E_DIM * WROW];
    __shared__ float  sScore[TOK_BLK * SROW];

    const int tid    = threadIdx.x;
    const int lane   = tid & 31;          // wave32 on gfx1250
    const int waveId = tid >> 5;          // 8 waves
    const int l16    = lane & 15;
    const bool hiHalf = (lane >= 16);

    const int blockBase = blockIdx.x * TOK_BLK;        // first token of block
    const int tokBase   = blockBase + waveId * 16;     // this wave's 16 tokens

    // per-lane x row pointer: row m = tokBase + (lane%16); lanes 16..31 read
    // the same rows at k+16 so each pair of half-waves covers 128B contiguously.
    const float* __restrict__ xr = x + (size_t)(tokBase + l16) * D_DIM;
    const int kkBias = hiHalf ? 16 : 0;   // A-layout K offset per half-wave

    v8f acc0 = {}, acc1 = {}, acc2 = {}, acc3 = {};

    for (int kp = 0; kp < D_DIM; kp += KPANEL) {
        __syncthreads();   // previous panel fully consumed
        // ---- stage W[0:64, kp:kp+KPANEL) as bf16 into LDS ----
        // phase 1: 8 independent coalesced global b128 loads (clause, 1 wait)
        v4f w4[8];
        #pragma unroll
        for (int j = 0; j < 8; ++j) {
            const int i = tid + j * 256;         // float4 index
            const int e = i >> 5;                // 32 float4s per 128-col row
            const int kq = (i & 31) << 2;
            w4[j] = *(const v4f*)(W + (size_t)e * D_DIM + kp + kq);
        }
        // phase 2: pack to bf16 and ds_store_b64
        #pragma unroll
        for (int j = 0; j < 8; ++j) {
            const int i = tid + j * 256;
            const int e = i >> 5;
            const int kq = (i & 31) << 2;
            v4bf hv;
            hv.x = (__bf16)w4[j].x; hv.y = (__bf16)w4[j].y;
            hv.z = (__bf16)w4[j].z; hv.w = (__bf16)w4[j].w;
            *(v4bf*)(&sW[e * WROW + kq]) = hv;   // 272B row stride, 8B aligned
        }
        __syncthreads();

        #pragma unroll
        for (int ks = 0; ks < KPANEL; ks += 32) {
            const int k0 = kp + ks;
            const int kk = k0 + kkBias;

            // ---- A fragment: 16 bf16 per lane, contiguous K layout
            //  halfs 0..15 = K kk..kk+15 ----
            v4f f0 = __builtin_nontemporal_load((const v4f*)(xr + kk));
            v4f f1 = __builtin_nontemporal_load((const v4f*)(xr + kk + 4));
            v4f f2 = __builtin_nontemporal_load((const v4f*)(xr + kk + 8));
            v4f f3 = __builtin_nontemporal_load((const v4f*)(xr + kk + 12));
            // prefetch next panel of this row (gfx1250 global_prefetch_b8)
            if (k0 + KPANEL < D_DIM)
                __builtin_prefetch(xr + kk + KPANEL, 0, 1);

            // ---- B fragments: hoist all 8 ds_load_b128 ahead of the WMMAs
            //  lane holds column e = tile*16+l16, K = ks+(hiHalf?16:0)..+15 ----
            const int kb = ks + (hiHalf ? 16 : 0);
            Bits256 raw0, raw1, raw2, raw3;
            {
                const __bf16* p = &sW[(0 * 16 + l16) * WROW + kb];
                raw0.lo = *(const v4u*)p; raw0.hi = *(const v4u*)(p + 8);
            }
            {
                const __bf16* p = &sW[(1 * 16 + l16) * WROW + kb];
                raw1.lo = *(const v4u*)p; raw1.hi = *(const v4u*)(p + 8);
            }
            {
                const __bf16* p = &sW[(2 * 16 + l16) * WROW + kb];
                raw2.lo = *(const v4u*)p; raw2.hi = *(const v4u*)(p + 8);
            }
            {
                const __bf16* p = &sW[(3 * 16 + l16) * WROW + kb];
                raw3.lo = *(const v4u*)p; raw3.hi = *(const v4u*)(p + 8);
            }

            v16bf a;
            a[0]=(__bf16)f0.x; a[1]=(__bf16)f0.y; a[2]=(__bf16)f0.z; a[3]=(__bf16)f0.w;
            a[4]=(__bf16)f1.x; a[5]=(__bf16)f1.y; a[6]=(__bf16)f1.z; a[7]=(__bf16)f1.w;
            a[8]=(__bf16)f2.x; a[9]=(__bf16)f2.y; a[10]=(__bf16)f2.z; a[11]=(__bf16)f2.w;
            a[12]=(__bf16)f3.x; a[13]=(__bf16)f3.y; a[14]=(__bf16)f3.z; a[15]=(__bf16)f3.w;

            acc0 = __builtin_amdgcn_wmma_f32_16x16x32_bf16(false, a, false,
                       __builtin_bit_cast(v16bf, raw0), (short)0, acc0, false, false);
            acc1 = __builtin_amdgcn_wmma_f32_16x16x32_bf16(false, a, false,
                       __builtin_bit_cast(v16bf, raw1), (short)0, acc1, false, false);
            acc2 = __builtin_amdgcn_wmma_f32_16x16x32_bf16(false, a, false,
                       __builtin_bit_cast(v16bf, raw2), (short)0, acc2, false, false);
            acc3 = __builtin_amdgcn_wmma_f32_16x16x32_bf16(false, a, false,
                       __builtin_bit_cast(v16bf, raw3), (short)0, acc3, false, false);
        }
    }

    // ---- epilogue from accumulators (C/D layout: N=l16, M=r(+8 hi half)) ----
    // write logits (+bias) coalesced; compute gumbel scores into LDS.
    const int mOff = hiHalf ? 8 : 0;
    #pragma unroll
    for (int tile = 0; tile < 4; ++tile) {
        const int e = tile * 16 + l16;
        const float bias = b[e];
        const v8f acc = (tile == 0) ? acc0 : (tile == 1) ? acc1 : (tile == 2) ? acc2 : acc3;
        #pragma unroll
        for (int r = 0; r < 8; ++r) {
            const int mLoc = waveId * 16 + r + mOff;
            const size_t t = (size_t)(blockBase + mLoc);
            const float logit = acc[r] + bias;
            __builtin_nontemporal_store(logit, &logitsOut[t * E_DIM + e]);
            const float uv = __builtin_nontemporal_load(&u[t * E_DIM + e]);
            const float g  = -__logf(-__logf(uv + EPS_F) + EPS_F);
            sScore[mLoc * SROW + e] = logit + g;   // TEMPERATURE == 1
        }
    }
    __syncthreads();

    // ---- per-token top-2 + softmax-of-2 (128 active threads) ----
    if (tid < TOK_BLK) {
        const int mLoc = tid;
        const size_t t = (size_t)(blockBase + mLoc);
        float s0 = -3.402823466e+38f, s1 = -3.402823466e+38f;
        int   i0 = 0, i1 = 0;
        #pragma unroll
        for (int e = 0; e < E_DIM; ++e) {
            const float v = sScore[mLoc * SROW + e];
            if (v > s0)       { s1 = s0; i1 = i0; s0 = v; i0 = e; }
            else if (v > s1)  { s1 = v; i1 = e; }
        }
        const float w0 = 1.0f / (1.0f + __expf(s1 - s0));
        idxOut[t * 2 + 0] = i0;
        idxOut[t * 2 + 1] = i1;
        wOut[t * 2 + 0] = w0;
        wOut[t * 2 + 1] = 1.0f - w0;
    }
}

extern "C" void kernel_launch(void* const* d_in, const int* in_sizes, int n_in,
                              void* d_out, int out_size, void* d_ws, size_t ws_size,
                              hipStream_t stream) {
    const float* x = (const float*)d_in[0];   // [T, D]
    const float* W = (const float*)d_in[1];   // [E, D]
    const float* b = (const float*)d_in[2];   // [E]
    const float* u = (const float*)d_in[3];   // [T, E]

    const int E = in_sizes[2];                // 64
    const int D = in_sizes[1] / E;            // 1024
    const int T = in_sizes[0] / D;            // 262144

    // output layout: indices [T,2] int32 | logits [T,E] f32 | weights [T,2] f32
    int*   idxOut    = (int*)d_out;
    float* logitsOut = (float*)d_out + (size_t)T * 2;
    float* wOut      = logitsOut + (size_t)T * E;

    const int blocks = T / TOK_BLK;           // 262144 / 128 = 2048
    router_gate_kernel<<<blocks, 256, 0, stream>>>(x, W, b, u,
                                                   idxOut, logitsOut, wOut, T);
}
